// LLMMamba_36301063586178
// MI455X (gfx1250) — matrix operand order repaired and checked by reading, not compiled, not run on hardware
//
#include <hip/hip_runtime.h>
#include <stdint.h>

#define NUM_LAYERS 4
#define D_IN 768
#define D_M 1536
#define D_S 16
#define D_D 96
#define KCONV 4
#define BATCH 2
#define SEQLEN 1024
#define ROWS (BATCH * SEQLEN)

typedef __attribute__((ext_vector_type(16))) __bf16   v16bf;
typedef __attribute__((ext_vector_type(8)))  float    v8f;
typedef __attribute__((ext_vector_type(4)))  uint32_t u32x4;
typedef __attribute__((ext_vector_type(8)))  uint32_t u32x8;

__device__ __forceinline__ unsigned short f2bf(float f) {
  uint32_t u = __float_as_uint(f);
  uint32_t r = u + 0x7FFFu + ((u >> 16) & 1u);   // round-to-nearest-even
  return (unsigned short)(r >> 16);
}

// ---------------------------------------------------------------------------
// Tensor Data Mover: DMA a (rows x 32) bf16 tile (row stride = row_stride_elems)
// from global into LDS at lds_byte_off, padded +4 dwords per 16 dwords so the
// LDS row stride is 40 elements (80 B, 16B-aligned rows). OOB rows/cols are
// zero-filled by the TDM. The LDS pointer is passed as an (unused) asm operand
// purely so its address escapes and alias analysis keeps the LDS reads live.
// ---------------------------------------------------------------------------
__device__ __forceinline__ void tdm_load_tile(uint32_t lds_byte_off,
                                              void* lds_base,
                                              const unsigned short* gptr,
                                              uint32_t rem_k_elems,
                                              uint32_t rem_rows,
                                              uint32_t row_stride_elems) {
  uint64_t ga = (uint64_t)(uintptr_t)gptr;
  u32x4 g0;
  g0[0] = 1u;                                            // count = 1 (valid D#)
  g0[1] = lds_byte_off;                                  // lds_addr
  g0[2] = (uint32_t)ga;                                  // global_addr[31:0]
  g0[3] = (uint32_t)((ga >> 32) & 0x1FFFFFFu) | (2u << 30); // addr[56:32] | type=2
  u32x8 g1;
  g1[0] = (1u << 16)    // data_size = 2 bytes
        | (1u << 20)    // pad_enable
        | (3u << 22)    // pad_interval = 16 dwords
        | (3u << 25);   // pad_amount  = 4 dwords  -> LDS row stride 40 elems
  g1[1] = (rem_k_elems & 0xFFFFu) << 16;                 // tensor_dim0[15:0]
  g1[2] = (rem_k_elems >> 16) | ((rem_rows & 0xFFFFu) << 16); // dim0 hi | dim1 lo
  g1[3] = (rem_rows >> 16) | (32u << 16);                // dim1 hi | tile_dim0 = 32
  g1[4] = 128u;                                          // tile_dim1 = 128, tile_dim2 = 0
  g1[5] = row_stride_elems;                              // tensor_dim0_stride[31:0]
  g1[6] = 0u;
  g1[7] = 0u;
  u32x4 gz = {};
  asm volatile("tensor_load_to_lds %0, %1, %2, %3"
               :: "s"(g0), "s"(g1), "s"(gz), "s"(gz), "v"(lds_base)
               : "memory");
}

// ---------------------------------------------------------------------------
// WMMA GEMM: C[M,N] (f32, row-major) = A[M,K](bf16) * Bw[N,K](bf16)^T
//            (+ bias[n]) (+ residual[m,n]); optional bf16 copy of C.
// Block = 256 threads = 8 waves, 128x128 tile; wave = 64x32 (8 accumulators).
// Double-buffered TDM staging overlapped with v_wmma_f32_16x16x32_bf16 math.
// ---------------------------------------------------------------------------
#define BT_K 32
#define LDS_STRIDE 40
#define TILE_HALF (128 * LDS_STRIDE)       // elems per A or B tile

__global__ __launch_bounds__(256)
void wmma_gemm_bf16(const unsigned short* __restrict__ A,
                    const unsigned short* __restrict__ Bw,
                    float* __restrict__ C,
                    unsigned short* __restrict__ Cbf,
                    const float* __restrict__ residual,
                    const float* __restrict__ bias,
                    int M, int N, int Kdim) {
  __shared__ __align__(16) unsigned short lds[2 * 2 * TILE_HALF]; // [buf][A|B]

  const int m0   = blockIdx.y * 128;
  const int n0   = blockIdx.x * 128;
  const int tid  = threadIdx.x;
  const int lane = tid & 31;
  const int wv   = tid >> 5;
  const int wy   = wv >> 2;        // 0..1  (64 rows each)
  const int wx   = wv & 3;         // 0..3  (32 cols each)
  const int hlf  = lane >> 4;
  const int l15  = lane & 15;

  v8f acc[4][2] = {};

  const int nk = Kdim / BT_K;

  // prologue: DMA tile 0 into buffer 0
  if (tid < 32) {
    tdm_load_tile(0u, lds, A + (size_t)m0 * Kdim,
                  (uint32_t)Kdim, (uint32_t)(M - m0), (uint32_t)Kdim);
    tdm_load_tile((uint32_t)(TILE_HALF * 2), lds, Bw + (size_t)n0 * Kdim,
                  (uint32_t)Kdim, (uint32_t)(N - n0), (uint32_t)Kdim);
  }

  for (int it = 0; it < nk; ++it) {
    const int cur = it & 1;
    if (tid < 32)
      __builtin_amdgcn_s_wait_tensorcnt(0);  // buf[cur] DMA complete
    __syncthreads();                         // visible; prior readers of buf[cur^1] done

    if (tid < 32 && (it + 1) < nk) {         // prefetch next tile into other buffer
      const int k1 = (it + 1) * BT_K;
      tdm_load_tile((uint32_t)((cur ^ 1) * 2 * TILE_HALF * 2), lds,
                    A + (size_t)m0 * Kdim + k1,
                    (uint32_t)(Kdim - k1), (uint32_t)(M - m0), (uint32_t)Kdim);
      tdm_load_tile((uint32_t)(((cur ^ 1) * 2 + 1) * TILE_HALF * 2), lds,
                    Bw + (size_t)n0 * Kdim + k1,
                    (uint32_t)(Kdim - k1), (uint32_t)(N - n0), (uint32_t)Kdim);
    }

    const unsigned short* ldsA = lds + (size_t)cur * 2 * TILE_HALF;
    const unsigned short* ldsB = ldsA + TILE_HALF;

    union Frag { v16bf v; uint4 q[2]; };
    Frag afr[4], bfr[2];
    #pragma unroll
    for (int fm = 0; fm < 4; fm++) {         // A 16x32 bf16 lane layout
      const unsigned short* base =
          &ldsA[(wy * 64 + fm * 16 + l15) * LDS_STRIDE + hlf * 8];
      afr[fm].q[0] = *(const uint4*)(base);       // K = hlf*8 + 0..7
      afr[fm].q[1] = *(const uint4*)(base + 16);  // K = hlf*8 + 16..23
    }
    #pragma unroll
    for (int fn = 0; fn < 2; fn++) {         // B 32x16 bf16 lane layout
      const unsigned short* base =
          &ldsB[(wx * 32 + fn * 16 + l15) * LDS_STRIDE + hlf * 8];
      bfr[fn].q[0] = *(const uint4*)(base);       // K = hlf*8 + 0..7
      bfr[fn].q[1] = *(const uint4*)(base + 16);  // K = hlf*8 + 16..23
    }
    #pragma unroll
    for (int fm = 0; fm < 4; fm++)
      #pragma unroll
      for (int fn = 0; fn < 2; fn++)
        acc[fm][fn] = __builtin_amdgcn_wmma_f32_16x16x32_bf16(
            false, afr[fm].v, false, bfr[fn].v, (short)0, acc[fm][fn],
            false, false);
  }

  #pragma unroll
  for (int fm = 0; fm < 4; fm++)
    #pragma unroll
    for (int fn = 0; fn < 2; fn++)
      #pragma unroll
      for (int j = 0; j < 8; j++) {          // C/D: m = j + 8*hlf, n = l15
        int cm = m0 + wy * 64 + fm * 16 + hlf * 8 + j;
        int cn = n0 + wx * 32 + fn * 16 + l15;
        if (cm < M && cn < N) {
          float v = acc[fm][fn][j];
          if (bias)     v += bias[cn];
          if (residual) v += residual[(size_t)cm * N + cn];
          C[(size_t)cm * N + cn] = v;
          if (Cbf) Cbf[(size_t)cm * N + cn] = f2bf(v);
        }
      }
}

// ---------------------------------------------------------------------------
// Elementwise / helper kernels
// ---------------------------------------------------------------------------
__global__ void cvt_bf16_kernel(const float* __restrict__ s,
                                unsigned short* __restrict__ d, int n) {
  int i = blockIdx.x * blockDim.x + threadIdx.x;
  if (i < n) d[i] = f2bf(s[i]);
}

__global__ void embed_kernel(const int* __restrict__ tok,
                             const float* __restrict__ emb,
                             float* __restrict__ seq) {
  int idx = blockIdx.x * blockDim.x + threadIdx.x;
  if (idx >= ROWS * D_IN) return;
  int r = idx / D_IN, c = idx % D_IN;
  seq[idx] = emb[(size_t)tok[r] * D_IN + c];
}

__global__ __launch_bounds__(256)
void rmsnorm_kernel(const float* __restrict__ x, const float* __restrict__ w,
                    unsigned short* __restrict__ out, int cols) {
  __shared__ float red[256];
  int row = blockIdx.x;
  const float* xr = x + (size_t)row * cols;
  float ss = 0.0f;
  for (int c = threadIdx.x; c < cols; c += 256) { float v = xr[c]; ss += v * v; }
  red[threadIdx.x] = ss;
  __syncthreads();
  for (int o = 128; o > 0; o >>= 1) {
    if (threadIdx.x < o) red[threadIdx.x] += red[threadIdx.x + o];
    __syncthreads();
  }
  float scale = rsqrtf(red[0] / (float)cols + 1e-6f);
  for (int c = threadIdx.x; c < cols; c += 256)
    out[(size_t)row * cols + c] = f2bf(xr[c] * scale * w[c]);
}

__global__ void conv_silu_kernel(const float* __restrict__ ab,
                                 const float* __restrict__ cw,
                                 const float* __restrict__ cb,
                                 float* __restrict__ af,
                                 unsigned short* __restrict__ abf) {
  int idx = blockIdx.x * blockDim.x + threadIdx.x;
  if (idx >= ROWS * D_M) return;
  int d = idx % D_M;
  int r = idx / D_M;
  int b = r >> 10, t = r & 1023;
  float acc = cb[d];
  #pragma unroll
  for (int j = 0; j < KCONV; j++) {
    int tt = t - (KCONV - 1) + j;
    if (tt >= 0)
      acc += cw[d * KCONV + j] * ab[((size_t)(b * SEQLEN + tt)) * (2 * D_M) + d];
  }
  float s = acc / (1.0f + __expf(-acc));   // SiLU
  af[idx]  = s;
  abf[idx] = f2bf(s);
}

// Selective scan: lane = one channel, 16-state recurrence in registers,
// Bm/Cm rows broadcast through LDS in T-chunks of 64.
__global__ __launch_bounds__(256)
void scan_kernel(const float* __restrict__ af,     // (R, D_M)  SiLU'd conv out
                 const float* __restrict__ dpre,   // (R, D_M)  pre-softplus delta
                 const float* __restrict__ Bm,     // (R, 16)
                 const float* __restrict__ Cm,     // (R, 16)
                 const float* __restrict__ ab,     // (R, 2*D_M) for bgate
                 const float* __restrict__ A_log,  // (D_M, 16)
                 const float* __restrict__ Dp,     // (D_M)
                 unsigned short* __restrict__ ybf) // (R, D_M)
{
  const int TB = 64;
  __shared__ float sB[TB][D_S];
  __shared__ float sC[TB][D_S];
  int d = blockIdx.x * blockDim.x + threadIdx.x;   // 0..D_M-1
  int b = blockIdx.y;

  float Ac[D_S];
  #pragma unroll
  for (int s = 0; s < D_S; s++) Ac[s] = -__expf(A_log[d * D_S + s]);
  float dcoef = Dp[d];
  float h[D_S];
  #pragma unroll
  for (int s = 0; s < D_S; s++) h[s] = 0.0f;

  for (int tc = 0; tc < SEQLEN; tc += TB) {
    __syncthreads();
    for (int i = threadIdx.x; i < TB * D_S; i += blockDim.x) {
      int ti = i / D_S, si = i % D_S;
      size_t rid = (size_t)(b * SEQLEN + tc + ti);
      sB[ti][si] = Bm[rid * D_S + si];
      sC[ti][si] = Cm[rid * D_S + si];
    }
    __syncthreads();
    for (int i = 0; i < TB; i++) {
      size_t rid = (size_t)(b * SEQLEN + tc + i);
      float a  = af[rid * D_M + d];
      float dp = dpre[rid * D_M + d];
      float delta = (dp > 20.0f) ? dp : log1pf(__expf(dp));   // softplus
      float da = delta * a;
      float out = 0.0f;
      #pragma unroll
      for (int s = 0; s < D_S; s++) {
        h[s] = __expf(delta * Ac[s]) * h[s] + sB[i][s] * da;
        out += h[s] * sC[i][s];
      }
      out += dcoef * a;
      float g = ab[rid * (2 * D_M) + D_M + d];
      float y = out * (g / (1.0f + __expf(-g)));
      ybf[rid * D_M + d] = f2bf(y);
    }
  }
}

// ---------------------------------------------------------------------------
// Host orchestration
// ---------------------------------------------------------------------------
extern "C" void kernel_launch(void* const* d_in, const int* in_sizes, int n_in,
                              void* d_out, int out_size, void* d_ws, size_t ws_size,
                              hipStream_t stream) {
  const int*   tok        = (const int*)  d_in[0];
  const float* embedding  = (const float*)d_in[1];
  const float* in_proj_w  = (const float*)d_in[2];
  const float* out_proj_w = (const float*)d_in[3];
  const float* sB_w       = (const float*)d_in[4];
  const float* sC_w       = (const float*)d_in[5];
  const float* sD0_w      = (const float*)d_in[6];
  const float* sD1_w      = (const float*)d_in[7];
  const float* sD1_b      = (const float*)d_in[8];
  const float* conv_w     = (const float*)d_in[9];
  const float* conv_b     = (const float*)d_in[10];
  const float* A_log      = (const float*)d_in[11];
  const float* Dp         = (const float*)d_in[12];
  const float* norm_w     = (const float*)d_in[13];
  const float* norm_f_w   = (const float*)d_in[14];
  const float* head_w     = (const float*)d_in[15];

  char*  ws  = (char*)d_ws;
  size_t off = 0;
  auto alloc = [&](size_t bytes) -> void* {
    void* p = ws + off;
    off = (off + bytes + 255) & ~(size_t)255;
    return p;
  };

  unsigned short* w_in_bf   = (unsigned short*)alloc((size_t)NUM_LAYERS * 2 * D_M * D_IN * 2);
  unsigned short* w_out_bf  = (unsigned short*)alloc((size_t)NUM_LAYERS * D_IN * D_M * 2);
  unsigned short* w_sB_bf   = (unsigned short*)alloc((size_t)NUM_LAYERS * D_S * D_M * 2);
  unsigned short* w_sC_bf   = (unsigned short*)alloc((size_t)NUM_LAYERS * D_S * D_M * 2);
  unsigned short* w_sD0_bf  = (unsigned short*)alloc((size_t)NUM_LAYERS * D_D * D_M * 2);
  unsigned short* w_sD1_bf  = (unsigned short*)alloc((size_t)NUM_LAYERS * D_M * D_D * 2);
  unsigned short* w_head_bf = (unsigned short*)alloc((size_t)32000 * D_IN * 2);

  float*          seq_f   = (float*)         alloc((size_t)ROWS * D_IN * 4);
  unsigned short* xn_bf   = (unsigned short*)alloc((size_t)ROWS * D_IN * 2);
  float*          ab_f    = (float*)         alloc((size_t)ROWS * 2 * D_M * 4);
  float*          aact_f  = (float*)         alloc((size_t)ROWS * D_M * 4);
  unsigned short* aact_bf = (unsigned short*)alloc((size_t)ROWS * D_M * 2);
  float*          Bm_f    = (float*)         alloc((size_t)ROWS * D_S * 4);
  float*          Cm_f    = (float*)         alloc((size_t)ROWS * D_S * 4);
  float*          dd_f    = (float*)         alloc((size_t)ROWS * D_D * 4);
  unsigned short* dd_bf   = (unsigned short*)alloc((size_t)ROWS * D_D * 2);
  float*          dpre_f  = (float*)         alloc((size_t)ROWS * D_M * 4);
  unsigned short* y_bf    = (unsigned short*)alloc((size_t)ROWS * D_M * 2);

  auto cvt = [&](const float* s, unsigned short* d, int n) {
    cvt_bf16_kernel<<<(n + 255) / 256, 256, 0, stream>>>(s, d, n);
  };
  auto gemm = [&](const unsigned short* A, const unsigned short* Bw, float* C,
                  unsigned short* Cbf, const float* res, const float* bias,
                  int M, int N, int K) {
    dim3 g((N + 127) / 128, (M + 127) / 128);
    wmma_gemm_bf16<<<g, 256, 0, stream>>>(A, Bw, C, Cbf, res, bias, M, N, K);
  };

  // one-time (per launch) bf16 weight conversion
  cvt(in_proj_w,  w_in_bf,   NUM_LAYERS * 2 * D_M * D_IN);
  cvt(out_proj_w, w_out_bf,  NUM_LAYERS * D_IN * D_M);
  cvt(sB_w,       w_sB_bf,   NUM_LAYERS * D_S * D_M);
  cvt(sC_w,       w_sC_bf,   NUM_LAYERS * D_S * D_M);
  cvt(sD0_w,      w_sD0_bf,  NUM_LAYERS * D_D * D_M);
  cvt(sD1_w,      w_sD1_bf,  NUM_LAYERS * D_M * D_D);
  cvt(head_w,     w_head_bf, 32000 * D_IN);

  embed_kernel<<<(ROWS * D_IN + 255) / 256, 256, 0, stream>>>(tok, embedding, seq_f);

  for (int i = 0; i < NUM_LAYERS; i++) {
    rmsnorm_kernel<<<ROWS, 256, 0, stream>>>(seq_f, norm_w + (size_t)i * D_IN, xn_bf, D_IN);

    gemm(xn_bf, w_in_bf + (size_t)i * 2 * D_M * D_IN, ab_f, nullptr, nullptr, nullptr,
         ROWS, 2 * D_M, D_IN);

    conv_silu_kernel<<<(ROWS * D_M + 255) / 256, 256, 0, stream>>>(
        ab_f, conv_w + (size_t)i * D_M * KCONV, conv_b + (size_t)i * D_M,
        aact_f, aact_bf);

    gemm(aact_bf, w_sB_bf  + (size_t)i * D_S * D_M, Bm_f, nullptr, nullptr, nullptr,
         ROWS, D_S, D_M);
    gemm(aact_bf, w_sC_bf  + (size_t)i * D_S * D_M, Cm_f, nullptr, nullptr, nullptr,
         ROWS, D_S, D_M);
    gemm(aact_bf, w_sD0_bf + (size_t)i * D_D * D_M, dd_f, dd_bf, nullptr, nullptr,
         ROWS, D_D, D_M);
    gemm(dd_bf,   w_sD1_bf + (size_t)i * D_M * D_D, dpre_f, nullptr, nullptr,
         sD1_b + (size_t)i * D_M, ROWS, D_M, D_D);

    scan_kernel<<<dim3(D_M / 256, BATCH), 256, 0, stream>>>(
        aact_f, dpre_f, Bm_f, Cm_f, ab_f,
        A_log + (size_t)i * D_M * D_S, Dp + (size_t)i * D_M, y_bf);

    // out = y @ out_proj^T + residual(seq) -> seq  (aliased read-then-write per element)
    gemm(y_bf, w_out_bf + (size_t)i * D_IN * D_M, seq_f, nullptr, seq_f, nullptr,
         ROWS, D_IN, D_M);
  }

  rmsnorm_kernel<<<ROWS, 256, 0, stream>>>(seq_f, norm_f_w, xn_bf, D_IN);
  gemm(xn_bf, w_head_bf, (float*)d_out, nullptr, nullptr, nullptr, ROWS, 32000, D_IN);

  (void)in_sizes; (void)n_in; (void)out_size; (void)ws_size;
}
